// cofeature_fast_56212531970449
// MI455X (gfx1250) — matrix operand order repaired and checked by reading, not compiled, not run on hardware
//
#include <hip/hip_runtime.h>
#include <hip/hip_bf16.h>

typedef __attribute__((ext_vector_type(16))) _Float16 v16h;
typedef __attribute__((ext_vector_type(8)))  float    v8f;

#define B_    8
#define C_    32
#define H_    256
#define W_    256
#define NOFF  5
#define ROWS  4
#define WCHUNK 64
#define TCOLS (WCHUNK + 2)   // 66 staged columns (chunk + halo)
#define TPITCH 68            // padded pitch
#define TELEMS (2 * C_ * TPITCH)
#define NIT   (ROWS * (W_ / WCHUNK))   // 16 chunk iterations per workgroup

// -------- zero the workspace accumulators (re-zeroed every launch) --------
__global__ void cofe_zero(float* __restrict__ p, int n) {
    int i = blockIdx.x * blockDim.x + threadIdx.x;
    if (i < n) p[i] = 0.0f;
}

// Issue async global->LDS copies for one (row h, chunk x0) tile.
// GVS mode: saddr = per-batch base (SGPR pair), vaddr = signed 32-bit byte offset.
__device__ __forceinline__ void stage_async(const float* __restrict__ xb,
                                            float* __restrict__ dstbase,
                                            int h, int x0, int tid) {
    for (int e = tid; e < 2 * C_ * TCOLS; e += 160) {
        int r   = e / (C_ * TCOLS);
        int rem = e - r * (C_ * TCOLS);
        int c   = rem / TCOLS;
        int j   = rem - c * TCOLS;
        int gh  = h + (r - 1); gh = gh < 0 ? 0 : gh;
        int gw  = x0 - 1 + j;  gw = gw < 0 ? 0 : (gw > W_ - 1 ? W_ - 1 : gw);
        int goff = ((c * H_ + gh) * W_ + gw) * 4;     // < 8.4MB, fits i32
        // flat LDS pointers carry the LDS offset in addr[31:0]
        unsigned laddr = (unsigned)(uintptr_t)(dstbase + (r * C_ + c) * TPITCH + j);
        asm volatile("global_load_async_to_lds_b32 %0, %1, %2"
                     :: "v"(laddr), "v"(goff), "s"(xb)
                     : "memory");
    }
}

// -------- main: per (batch, 4-row strip); 5 waves = 5 offsets --------
__global__ __launch_bounds__(160) void cofe_main(const float* __restrict__ x,
                                                 float* __restrict__ acc) {
    __shared__ float tile[2][TELEMS];         // double-buffered: 2 x (2 rows x 32ch x 68col)
    __shared__ float simbuf[NOFF * 32];

    const int b    = blockIdx.y;
    const int h0   = blockIdx.x * ROWS;
    const int tid  = threadIdx.x;
    const int o    = tid >> 5;                // wave id == offset id (0..4)
    const int lane = tid & 31;
    const int dy   = (o < 3) ? -1 : 0;        // offsets: (-1,-1)(-1,0)(-1,1)(0,-1)(0,0)
    const int dx   = (o < 3) ? (o - 1) : (o - 4);
    const int rs   = 1 + dy;                  // side row inside the tile
    const int hi   = lane >> 4;
    const int lr   = lane & 15;

    v8f acc00 = {}; v8f acc01 = {}; v8f acc10 = {}; v8f acc11 = {};

    const float* xb = x + (size_t)b * C_ * H_ * W_;

    // prefetch first chunk
    stage_async(xb, &tile[0][0], h0, 0, tid);

    for (int it = 0; it < NIT; ++it) {
        // my async issues (targeting tile[it&1]) complete, then block-wide barrier:
        // every wave has waited on its own ASYNCcnt, so the whole tile is visible.
        asm volatile("s_wait_asynccnt 0x0" ::: "memory");
        __syncthreads();

        // prefetch the next chunk into the other buffer (safe: all waves finished
        // reading it before this barrier)
        if (it + 1 < NIT) {
            const int nit = it + 1;
            stage_async(xb, &tile[nit & 1][0], h0 + (nit >> 2),
                        (nit & 3) * WCHUNK, tid);
        }

        const float* tl = &tile[it & 1][0];

        #pragma unroll
        for (int kk = 0; kk < 2; ++kk) {      // two 32-pixel sub-chunks
            const int base = 1 + kk * 32;

            // per-pixel cosine similarity (pixel == lane)
            {
                float cn2 = 0.f, sn2 = 0.f, dt = 0.f;
                const int ccol = base + lane;
                const int scol = ccol + dx;
                #pragma unroll
                for (int c = 0; c < C_; ++c) {
                    float cv = tl[(C_ + c) * TPITCH + ccol];
                    float sv = tl[(rs * C_ + c) * TPITCH + scol];
                    cn2 = fmaf(cv, cv, cn2);
                    sn2 = fmaf(sv, sv, sn2);
                    dt  = fmaf(cv, sv, dt);
                }
                float s = (dt > 0.f) ? dt * rsqrtf(fmaxf(cn2 * sn2, 1e-30f)) : 0.f;
                simbuf[o * 32 + lane] = s;    // same-wave DS ops are in-order
            }

            // build WMMA fragments per the CDNA5 16-bit A/B layouts
            v16h a0, a1, b0, b1;
            #pragma unroll
            for (int e = 0; e < 16; ++e) {
                // A: lane half selects K subset: K = (e<8?0:16) + hi*8 + (e%8)
                int K   = ((e & 8) << 1) + (hi << 3) + (e & 7);
                float s = simbuf[o * 32 + K];
                int col = base + K;
                a0[e] = (_Float16)(tl[(C_ + lr)      * TPITCH + col] * s);
                a1[e] = (_Float16)(tl[(C_ + 16 + lr) * TPITCH + col] * s);
                // B: K = (e<8?0:16) + hi*8 + (e%8), N = lane%16
                int Kb   = ((e & 8) << 1) + (hi << 3) + (e & 7);
                int colb = base + Kb + dx;
                b0[e] = (_Float16)(tl[(rs * C_ + lr)      * TPITCH + colb]);
                b1[e] = (_Float16)(tl[(rs * C_ + 16 + lr) * TPITCH + colb]);
            }

            acc00 = __builtin_amdgcn_wmma_f32_16x16x32_f16(false, a0, false, b0,
                        (short)0, acc00, false, false);
            acc01 = __builtin_amdgcn_wmma_f32_16x16x32_f16(false, a0, false, b1,
                        (short)0, acc01, false, false);
            acc10 = __builtin_amdgcn_wmma_f32_16x16x32_f16(false, a1, false, b0,
                        (short)0, acc10, false, false);
            acc11 = __builtin_amdgcn_wmma_f32_16x16x32_f16(false, a1, false, b1,
                        (short)0, acc11, false, false);
        }
    }

    // D layout: VGPR v -> M = v + hi*8, N = lane%16; reduce across strips via atomics
    float* dst = acc + (size_t)(b * NOFF + o) * (C_ * C_);
    #pragma unroll
    for (int v = 0; v < 8; ++v) {
        int M = v + (hi << 3);
        atomicAdd(dst + (M)      * C_ + lr,      acc00[v]);
        atomicAdd(dst + (M)      * C_ + 16 + lr, acc01[v]);
        atomicAdd(dst + (16 + M) * C_ + lr,      acc10[v]);
        atomicAdd(dst + (16 + M) * C_ + 16 + lr, acc11[v]);
    }
}

// -------- normalize each (batch, offset) row of 1024 by its L2 norm --------
__global__ __launch_bounds__(256) void cofe_norm(const float* __restrict__ acc,
                                                 float* __restrict__ out) {
    __shared__ float red[256];
    const int row = blockIdx.x;               // 0..39
    const float* a = acc + (size_t)row * (C_ * C_);
    const int t = threadIdx.x;
    float s = 0.f;
    for (int i = t; i < C_ * C_; i += 256) { float v = a[i]; s = fmaf(v, v, s); }
    red[t] = s;
    __syncthreads();
    for (int off = 128; off > 0; off >>= 1) {
        if (t < off) red[t] += red[t + off];
        __syncthreads();
    }
    const float inv = 1.f / fmaxf(sqrtf(red[0]), 1e-12f);
    for (int i = t; i < C_ * C_; i += 256)
        out[(size_t)row * (C_ * C_) + i] = a[i] * inv;
}

extern "C" void kernel_launch(void* const* d_in, const int* in_sizes, int n_in,
                              void* d_out, int out_size, void* d_ws, size_t ws_size,
                              hipStream_t stream) {
    (void)in_sizes; (void)n_in; (void)out_size; (void)ws_size;
    const float* x = (const float*)d_in[0];
    float* out = (float*)d_out;
    float* ws  = (float*)d_ws;

    const int n_acc = B_ * NOFF * C_ * C_;    // 40960 floats of workspace
    cofe_zero<<<(n_acc + 255) / 256, 256, 0, stream>>>(ws, n_acc);

    dim3 grid(H_ / ROWS, B_);                 // 64 strips x 8 batches = 512 WGs
    cofe_main<<<grid, 160, 0, stream>>>(x, ws);

    cofe_norm<<<B_ * NOFF, 256, 0, stream>>>(ws, out);
}
